// PairEmbed_77970836291664
// MI455X (gfx1250) — matrix operand order repaired and checked by reading, not compiled, not run on hardware
//
#include <hip/hip_runtime.h>
#include <math.h>

#define DEV __device__ __forceinline__

typedef __attribute__((ext_vector_type(16))) _Float16 v16h;
typedef __attribute__((ext_vector_type(8)))  float    v8f;

constexpr int   B_    = 16, S_ = 256, D_ = 64;
constexpr int   P_    = S_ * (S_ - 1) / 2;      // 32640
constexpr int   NTOT  = B_ * P_;                 // 522240 columns
constexpr int   CHUNK = 32;                      // columns per block iteration (stats passes)
constexpr int   NCHUNK = NTOT / CHUNK;           // 16320
constexpr int   TI    = S_ / 16;                 // 16 tile-rows in (i,j) plane
constexpr int   NTILE_TRI = TI * (TI + 1) / 2;   // 136 triangular tiles per batch
constexpr float FEPS   = 1e-8f;
constexpr float BN_EPS = 1e-5f;

// ---- workspace layout (float offsets) ----
constexpr int OFS_T     = 0;                     // t[b*P+p] pair features (2 MB)
constexpr int OFS_STAT  = OFS_T + NTOT;          // [2] sum, sumsq of t
constexpr int OFS_A1    = OFS_STAT + 2;          // [64] stage-1 folded scale
constexpr int OFS_C1    = OFS_A1 + 64;           // [64] stage-1 folded shift
constexpr int OFS_SUMU  = OFS_C1 + 64;           // [64]
constexpr int OFS_GRAM2 = OFS_SUMU + 64;         // [64*64] sum u u^T
constexpr int OFS_S2    = OFS_GRAM2 + 4096;      // [64]
constexpr int OFS_C2    = OFS_S2 + 64;           // [64]
constexpr int OFS_SUMV  = OFS_C2 + 64;           // [64]
constexpr int OFS_GRAM3 = OFS_SUMV + 64;         // [64*64] sum v v^T
constexpr int OFS_S3    = OFS_GRAM3 + 4096;      // [64]
constexpr int OFS_C3    = OFS_S3 + 64;           // [64]
constexpr int OFS_W2H   = OFS_C3 + 64;           // 4096 halves = 2048 floats
constexpr int OFS_W3H   = OFS_W2H + 2048;        // 4096 halves
constexpr int ZERO_BEG  = OFS_STAT;
constexpr int ZERO_LEN  = OFS_W2H - OFS_STAT;    // accumulators to clear per call

DEV float gelu_exact(float x) { return 0.5f * x * (1.0f + erff(x * 0.70710678118654752f)); }

// A-matrix (16x32 f16) element->K map: lane = m + 16h, vector elem e
DEV int kaA(int e, int h) { return 16 * h + e; }

DEV v8f wmma32f16(v16h a, v16h b, v8f c) {
  return __builtin_amdgcn_wmma_f32_16x16x32_f16(false, a, false, b, (short)0, c, false, false);
}

DEV void pair_ij(int p, int& i, int& j) {
  i = (int)((1.0f + sqrtf(8.0f * (float)p + 1.0f)) * 0.5f);
  while (i * (i - 1) / 2 > p) --i;
  while ((i + 1) * i / 2 <= p) ++i;
  j = p - i * (i - 1) / 2;
}

// ---------- K0: clear accumulators ----------
__global__ void k0_zero(float* wsf) {
  int idx = blockIdx.x * 256 + threadIdx.x;
  if (idx < ZERO_LEN) wsf[ZERO_BEG + idx] = 0.0f;
}

// ---------- K1: pair features + moments ----------
__global__ __launch_bounds__(256) void k1_feat(const float* __restrict__ x, float* wsf) {
  int idx = blockIdx.x * 256 + threadIdx.x;
  float v = 0.0f, v2 = 0.0f;
  if (idx < NTOT) {
    int b = idx / P_;
    int p = idx - b * P_;
    int i, j; pair_ij(p, i, j);
    const float* xb = x + b * 4 * S_;
    float xi0 = xb[0 * S_ + i], xi1 = xb[1 * S_ + i], xi2 = xb[2 * S_ + i];
    float xj0 = xb[0 * S_ + j], xj1 = xb[1 * S_ + j], xj2 = xb[2 * S_ + j];
    float pi = fmaxf(sqrtf(xi0 * xi0 + xi1 * xi1 + xi2 * xi2), FEPS);
    float pj = fmaxf(sqrtf(xj0 * xj0 + xj1 * xj1 + xj2 * xj2), FEPS);
    float c = (xi0 * xj0 + xi1 * xj1 + xi2 * xj2) / (pi * pj);
    c = fminf(fmaxf(c, -1.0f), 1.0f);
    float delta = acosf(c);
    float t = logf(fmaxf(delta, FEPS));
    wsf[OFS_T + idx] = t;
    v = t; v2 = t * t;
  }
  __shared__ float rs[256], rq[256];
  int tid = threadIdx.x;
  rs[tid] = v; rq[tid] = v2;
  __syncthreads();
  for (int s = 128; s > 0; s >>= 1) {
    if (tid < s) { rs[tid] += rs[tid + s]; rq[tid] += rq[tid + s]; }
    __syncthreads();
  }
  if (tid == 0) { atomicAdd(&wsf[OFS_STAT], rs[0]); atomicAdd(&wsf[OFS_STAT + 1], rq[0]); }
}

// ---------- K2: fold bn0+w1+bn1 analytically; convert w2/w3 to f16 ----------
__global__ void k2_fold1(float* wsf,
                         const float* gamma0, const float* beta0,
                         const float* w1, const float* b1,
                         const float* g1, const float* be1,
                         const float* w2, const float* w3) {
  int d = threadIdx.x;  // 64 threads
  float invN = 1.0f / (float)NTOT;
  float mu0  = wsf[OFS_STAT] * invN;
  float var0 = fmaxf(wsf[OFS_STAT + 1] * invN - mu0 * mu0, 0.0f);
  float rs0  = rsqrtf(var0 + BN_EPS);
  float a0 = rs0 * gamma0[0];
  float b0 = beta0[0] - mu0 * a0;
  // bn0 output: tn = a0*t + b0  ->  exact moments: mean = a0*mu0+b0, var = a0^2*var0
  float mt = a0 * mu0 + b0;
  float vt = a0 * a0 * var0;
  float w1d = w1[d];
  float mean1 = w1d * mt + b1[d];
  float var1  = w1d * w1d * vt;
  float sg = rsqrtf(var1 + BN_EPS) * g1[d];
  wsf[OFS_A1 + d] = w1d * sg * a0;
  wsf[OFS_C1 + d] = w1d * sg * b0 + (b1[d] - mean1) * sg + be1[d];
  _Float16* w2h = (_Float16*)(wsf + OFS_W2H);
  _Float16* w3h = (_Float16*)(wsf + OFS_W3H);
  for (int k = 0; k < 64; ++k) {
    w2h[d * 64 + k] = (_Float16)w2[d * 64 + k];
    w3h[d * 64 + k] = (_Float16)w3[d * 64 + k];
  }
}

// ---------- K3: stage-2 stats: sum_u + Gram(u) via WMMA SYRK ----------
__global__ __launch_bounds__(256) void k3_stats2(float* wsf) {
  __shared__ _Float16 u_lds[64][CHUNK + 2];
  int tid = threadIdx.x;
  int lane = tid & 31, wave = tid >> 5;
  int n = lane & 15, h = lane >> 4;
  int mi  = wave >> 1;            // Gram tile row (tiles 2w, 2w+1)
  int ni0 = (wave & 1) * 2;       // Gram tile cols ni0, ni0+1
  v8f gA = {}, gB = {};
  float rowacc = 0.0f;
  for (int c = blockIdx.x; c < NCHUNK; c += gridDim.x) {
    int base = c * CHUNK;
    if (tid == 0 && c + (int)gridDim.x < NCHUNK)
      __builtin_prefetch(&wsf[OFS_T + (c + gridDim.x) * CHUNK], 0, 3);
    for (int e = tid; e < 64 * CHUNK; e += 256) {
      int d = e >> 5, col = e & 31;
      float t = wsf[OFS_T + base + col];
      u_lds[d][col] = (_Float16)gelu_exact(wsf[OFS_A1 + d] * t + wsf[OFS_C1 + d]);
    }
    __syncthreads();
    v16h a, b0, b1;
    for (int e = 0; e < 16; ++e) {
      a[e]  = u_lds[mi * 16 + n][kaA(e, h)];          // A: rows=channels, K=cols
      b0[e] = u_lds[ni0 * 16 + n][16 * h + e];        // B: K=cols, N=channels
      b1[e] = u_lds[(ni0 + 1) * 16 + n][16 * h + e];
    }
    gA = wmma32f16(a, b0, gA);
    gB = wmma32f16(a, b1, gB);
    if (tid < 64) {
      float s = 0.0f;
      for (int col = 0; col < CHUNK; ++col) s += (float)u_lds[tid][col];
      rowacc += s;
    }
    __syncthreads();
  }
  for (int r = 0; r < 8; ++r) {
    int m = mi * 16 + r + 8 * h;
    atomicAdd(&wsf[OFS_GRAM2 + m * 64 + ni0 * 16 + n], gA[r]);
    atomicAdd(&wsf[OFS_GRAM2 + m * 64 + (ni0 + 1) * 16 + n], gB[r]);
  }
  if (tid < 64) atomicAdd(&wsf[OFS_SUMU + tid], rowacc);
}

// ---------- K4/K6 shared: fold bn stats from Gram + channel sums ----------
DEV void fold_from_gram(float* wsf, int ofs_sum, int ofs_gram, int ofs_s, int ofs_c,
                        const float* w, const float* bb, const float* g, const float* be) {
  int d = threadIdx.x;  // 64 threads
  float invN = 1.0f / (float)NTOT;
  float mv = 0.0f;
  for (int k = 0; k < 64; ++k) mv += w[d * 64 + k] * wsf[ofs_sum + k];
  float mean = mv * invN + bb[d];
  float q = 0.0f;
  for (int k = 0; k < 64; ++k) {
    float acc = 0.0f;
    for (int l = 0; l < 64; ++l) acc += wsf[ofs_gram + k * 64 + l] * w[d * 64 + l];
    q += w[d * 64 + k] * acc;
  }
  float E2  = q * invN + 2.0f * bb[d] * mv * invN + bb[d] * bb[d];
  float var = fmaxf(E2 - mean * mean, 0.0f);
  float s = g[d] * rsqrtf(var + BN_EPS);
  wsf[ofs_s + d] = s;
  wsf[ofs_c + d] = be[d] - mean * s;
}

__global__ void k4_fold2(float* wsf, const float* w2, const float* b2,
                         const float* g2, const float* be2) {
  fold_from_gram(wsf, OFS_SUMU, OFS_GRAM2, OFS_S2, OFS_C2, w2, b2, g2, be2);
}
__global__ void k6_fold3(float* wsf, const float* w3, const float* b3,
                         const float* g3, const float* be3) {
  fold_from_gram(wsf, OFS_SUMV, OFS_GRAM3, OFS_S3, OFS_C3, w3, b3, g3, be3);
}

// ---------- K5: stage-3 stats: h2 = w2@u + b2 (WMMA GEMM), v=gelu(bn2), Gram(v) ----------
__global__ __launch_bounds__(256) void k5_stats3(float* wsf, const float* __restrict__ b2) {
  __shared__ _Float16 u_lds[64][CHUNK + 2];
  __shared__ _Float16 v_lds[64][CHUNK + 2];
  int tid = threadIdx.x;
  int lane = tid & 31, wave = tid >> 5;
  int n = lane & 15, h = lane >> 4;
  int mw = wave >> 1, nw = wave & 1;   // GEMM tile grid 4x2 over 64x32 panel
  int mi  = wave >> 1;                 // Gram tiles (same pairing as k3)
  int ni0 = (wave & 1) * 2;
  const _Float16* w2h = (const _Float16*)(wsf + OFS_W2H);
  v16h wa0, wa1;
  for (int e = 0; e < 16; ++e) {       // preload w2 A-fragments (K=64 -> 2 k-steps)
    wa0[e] = w2h[(mw * 16 + n) * 64 + kaA(e, h)];
    wa1[e] = w2h[(mw * 16 + n) * 64 + 32 + kaA(e, h)];
  }
  v8f cb;
  for (int r = 0; r < 8; ++r) cb[r] = b2[mw * 16 + r + 8 * h];
  v8f gA = {}, gB = {};
  float rowacc = 0.0f;
  for (int c = blockIdx.x; c < NCHUNK; c += gridDim.x) {
    int base = c * CHUNK;
    if (tid == 0 && c + (int)gridDim.x < NCHUNK)
      __builtin_prefetch(&wsf[OFS_T + (c + gridDim.x) * CHUNK], 0, 3);
    for (int e = tid; e < 64 * CHUNK; e += 256) {
      int d = e >> 5, col = e & 31;
      float t = wsf[OFS_T + base + col];
      u_lds[d][col] = (_Float16)gelu_exact(wsf[OFS_A1 + d] * t + wsf[OFS_C1 + d]);
    }
    __syncthreads();
    v16h bb0, bb1;
    for (int e = 0; e < 16; ++e) {
      bb0[e] = u_lds[16 * h + e][nw * 16 + n];
      bb1[e] = u_lds[32 + 16 * h + e][nw * 16 + n];
    }
    v8f hc = cb;
    hc = wmma32f16(wa0, bb0, hc);
    hc = wmma32f16(wa1, bb1, hc);
    for (int r = 0; r < 8; ++r) {
      int d = mw * 16 + r + 8 * h;
      v_lds[d][nw * 16 + n] = (_Float16)gelu_exact(wsf[OFS_S2 + d] * hc[r] + wsf[OFS_C2 + d]);
    }
    __syncthreads();
    v16h a, b0, b1;
    for (int e = 0; e < 16; ++e) {
      a[e]  = v_lds[mi * 16 + n][kaA(e, h)];
      b0[e] = v_lds[ni0 * 16 + n][16 * h + e];
      b1[e] = v_lds[(ni0 + 1) * 16 + n][16 * h + e];
    }
    gA = wmma32f16(a, b0, gA);
    gB = wmma32f16(a, b1, gB);
    if (tid < 64) {
      float s = 0.0f;
      for (int col = 0; col < CHUNK; ++col) s += (float)v_lds[tid][col];
      rowacc += s;
    }
    __syncthreads();
  }
  for (int r = 0; r < 8; ++r) {
    int m = mi * 16 + r + 8 * h;
    atomicAdd(&wsf[OFS_GRAM3 + m * 64 + ni0 * 16 + n], gA[r]);
    atomicAdd(&wsf[OFS_GRAM3 + m * 64 + (ni0 + 1) * 16 + n], gB[r]);
  }
  if (tid < 64) atomicAdd(&wsf[OFS_SUMV + tid], rowacc);
}

// ---------- K7: fused forward over 16x16 (i,j) tiles -> coalesced symmetric scatter ----------
// One block per (batch, triangular tile). N = 256 columns = one 16x16 tile of the pair plane.
// Both store orientations are emitted as 64B-contiguous runs via an LDS-transposed stage.
__global__ __launch_bounds__(256) void k7_out(float* wsf, const float* __restrict__ b2,
                                              const float* __restrict__ b3,
                                              float* __restrict__ out) {
  __shared__ _Float16 u_lds[64][264];
  __shared__ _Float16 v_lds[64][264];
  __shared__ float    y_lds[64][16][17];   // [d][di][dj], padded: conflict-free both ways

  int tid = threadIdx.x;
  int lane = tid & 31, wave = tid >> 5;
  int n = lane & 15, h = lane >> 4;
  int mw = wave & 3;                 // M-tile (channel group)
  int ng = (wave >> 2) * 8;          // N-tile group base (waves 0-3: 0..7, 4-7: 8..15)

  int bt = blockIdx.x / NTILE_TRI;
  int r  = blockIdx.x - bt * NTILE_TRI;
  int ti = (int)((sqrtf(8.0f * (float)r + 1.0f) - 1.0f) * 0.5f);   // r = ti(ti+1)/2 + tj
  while (ti * (ti + 1) / 2 > r) --ti;
  while ((ti + 1) * (ti + 2) / 2 <= r) ++ti;
  int tj = r - ti * (ti + 1) / 2;
  int I0 = ti * 16, J0 = tj * 16;
  bool diag = (ti == tj);

  // ---- stage u: one pair-column per thread, all 64 channels ----
  {
    int di = tid >> 4, dj = tid & 15;
    int i = I0 + di, j = J0 + dj;
    int p = (i > j) ? (i * (i - 1) / 2 + j) : 0;    // invalid cols: harmless dummy
    float t = wsf[OFS_T + bt * P_ + p];
    for (int d = 0; d < 64; ++d)
      u_lds[d][tid] = (_Float16)gelu_exact(wsf[OFS_A1 + d] * t + wsf[OFS_C1 + d]);
  }
  __syncthreads();

  const _Float16* w2h = (const _Float16*)(wsf + OFS_W2H);
  const _Float16* w3h = (const _Float16*)(wsf + OFS_W3H);
  v16h wa0, wa1, xa0, xa1;
  for (int e = 0; e < 16; ++e) {
    wa0[e] = w2h[(mw * 16 + n) * 64 + kaA(e, h)];
    wa1[e] = w2h[(mw * 16 + n) * 64 + 32 + kaA(e, h)];
    xa0[e] = w3h[(mw * 16 + n) * 64 + kaA(e, h)];
    xa1[e] = w3h[(mw * 16 + n) * 64 + 32 + kaA(e, h)];
  }
  v8f cb;
  for (int rr = 0; rr < 8; ++rr) cb[rr] = b2[mw * 16 + rr + 8 * h];

  // ---- GEMM w2: each wave does 8 N-tiles, one A-fragment pair reused ----
  v8f acc[8];
#pragma unroll
  for (int q = 0; q < 8; ++q) {
    int nt = ng + q;
    v16h bb0, bb1;
    for (int e = 0; e < 16; ++e) {
      bb0[e] = u_lds[16 * h + e][nt * 16 + n];
      bb1[e] = u_lds[32 + 16 * h + e][nt * 16 + n];
    }
    v8f hc = cb;
    hc = wmma32f16(wa0, bb0, hc);
    hc = wmma32f16(wa1, bb1, hc);
    acc[q] = hc;
  }
#pragma unroll
  for (int q = 0; q < 8; ++q) {
    for (int rr = 0; rr < 8; ++rr) {
      int d = mw * 16 + rr + 8 * h;
      v_lds[d][(ng + q) * 16 + n] =
          (_Float16)gelu_exact(wsf[OFS_S2 + d] * acc[q][rr] + wsf[OFS_C2 + d]);
    }
  }
  __syncthreads();

  // ---- GEMM w3 + bn3 fold -> y_lds ----
#pragma unroll
  for (int q = 0; q < 8; ++q) {
    int nt = ng + q;
    v16h bb0, bb1;
    for (int e = 0; e < 16; ++e) {
      bb0[e] = v_lds[16 * h + e][nt * 16 + n];
      bb1[e] = v_lds[32 + 16 * h + e][nt * 16 + n];
    }
    v8f y = {};
    y = wmma32f16(xa0, bb0, y);
    y = wmma32f16(xa1, bb1, y);
    for (int rr = 0; rr < 8; ++rr) {
      int d = mw * 16 + rr + 8 * h;
      float s3 = wsf[OFS_S3 + d];
      y_lds[d][nt][n] = s3 * (y[rr] + b3[d]) + wsf[OFS_C3 + d];  // col = nt*16+n -> di=nt, dj=n
    }
  }
  __syncthreads();

  // ---- store both orientations, 64B-contiguous runs ----
  // direct: lanes sweep dj  ->  out[b][d][I0+di][J0+dj] contiguous
  for (int k = 0; k < 64; ++k) {
    int e = tid + 256 * k;
    int dj = e & 15, di = (e >> 4) & 15, d = e >> 8;
    if (!diag || di > dj) {
      out[((bt * 64 + d) * 256 + I0 + di) * 256 + J0 + dj] = y_lds[d][di][dj];
    }
  }
  // mirrored: lanes sweep di  ->  out[b][d][J0+dj][I0+di] contiguous (LDS read conflict-free)
  for (int k = 0; k < 64; ++k) {
    int f = tid + 256 * k;
    int di = f & 15, dj = (f >> 4) & 15, d = f >> 8;
    if (!diag || di > dj) {
      out[((bt * 64 + d) * 256 + J0 + dj) * 256 + I0 + di] = y_lds[d][di][dj];
    }
  }
}

// ---------- K8: zero the diagonal ----------
__global__ void k8_diag(float* out) {
  int idx = blockIdx.x * 256 + threadIdx.x;
  if (idx >= B_ * D_ * S_) return;
  int i = idx & 255;
  int bd = idx >> 8;
  out[(bd * 256 + i) * 256 + i] = 0.0f;
}

extern "C" void kernel_launch(void* const* d_in, const int* in_sizes, int n_in,
                              void* d_out, int out_size, void* d_ws, size_t ws_size,
                              hipStream_t stream) {
  const float* x      = (const float*)d_in[0];
  const float* gamma0 = (const float*)d_in[1];
  const float* beta0  = (const float*)d_in[2];
  const float* w1  = (const float*)d_in[3];
  const float* b1  = (const float*)d_in[4];
  const float* g1  = (const float*)d_in[5];
  const float* be1 = (const float*)d_in[6];
  const float* w2  = (const float*)d_in[7];
  const float* b2  = (const float*)d_in[8];
  const float* g2  = (const float*)d_in[9];
  const float* be2 = (const float*)d_in[10];
  const float* w3  = (const float*)d_in[11];
  const float* b3  = (const float*)d_in[12];
  const float* g3  = (const float*)d_in[13];
  const float* be3 = (const float*)d_in[14];
  float* out = (float*)d_out;
  float* wsf = (float*)d_ws;

  k0_zero  <<<(ZERO_LEN + 255) / 256, 256, 0, stream>>>(wsf);
  k1_feat  <<<(NTOT + 255) / 256, 256, 0, stream>>>(x, wsf);
  k2_fold1 <<<1, 64, 0, stream>>>(wsf, gamma0, beta0, w1, b1, g1, be1, w2, w3);
  k3_stats2<<<1020, 256, 0, stream>>>(wsf);
  k4_fold2 <<<1, 64, 0, stream>>>(wsf, w2, b2, g2, be2);
  k5_stats3<<<1020, 256, 0, stream>>>(wsf, b2);
  k6_fold3 <<<1, 64, 0, stream>>>(wsf, w3, b3, g3, be3);
  k7_out   <<<B_ * NTILE_TRI, 256, 0, stream>>>(wsf, b2, b3, out);
  k8_diag  <<<(B_ * D_ * S_ + 255) / 256, 256, 0, stream>>>(out);
}
